// DoubleStreamBlock_43301860278750
// MI455X (gfx1250) — compile-verified, matrix operand order checked by reading, not run on hardware
//
#include <hip/hip_runtime.h>

// ---------------------------------------------------------------------------
// DoubleStreamBlock for MI455X (gfx1250): bf16 WMMA for all heavy matmuls,
// 128x128 GEMM tiles with double-buffered LDS, async global->LDS staging
// (ASYNCcnt) when the toolchain exposes the gfx1250 builtins.
// ---------------------------------------------------------------------------

constexpr int B_   = 2;
constexpr int LT_  = 256;
constexpr int LI_  = 1024;
constexpr int Ltot_= LT_ + LI_;     // 1280
constexpr int H_   = 3072;
constexpr int NH_  = 24;
constexpr int DH_  = 128;
constexpr int MLP_ = 4 * H_;        // 12288

typedef __attribute__((ext_vector_type(16))) __bf16 bf16x16;
typedef __attribute__((ext_vector_type(8)))  float  floatx8;

struct alignas(16) B128 { unsigned int w[4]; };

union Frag {
  bf16x16 v;
  B128    q[2];
  __device__ Frag() {}
};

__device__ inline unsigned short f32_to_bf16(float f) {
  unsigned int u = __float_as_uint(f);
  u += 0x7FFFu + ((u >> 16) & 1u);          // round-to-nearest-even
  return (unsigned short)(u >> 16);
}

__device__ inline floatx8 wmma_bf16(const Frag& a, const Frag& b, floatx8 c) {
  return __builtin_amdgcn_wmma_f32_16x16x32_bf16(
      /*neg_a=*/false, a.v, /*neg_b=*/false, b.v,
      /*c_mod=*/(short)0, c, /*reuse_a=*/false, /*reuse_b=*/false);
}

// --- gfx1250 async global->LDS copy (ASYNCcnt), with safe fallback ---------
#if defined(__has_builtin)
#if __has_builtin(__builtin_amdgcn_global_load_async_to_lds_b128) && \
    __has_builtin(__builtin_amdgcn_s_wait_asynccnt)
#define HAVE_ASYNC_LDS 1
#endif
#endif
#ifndef HAVE_ASYNC_LDS
#define HAVE_ASYNC_LDS 0
#endif

typedef int v4i_t __attribute__((vector_size(16)));

__device__ inline void cp16_g2l(const void* g, void* l) {
#if HAVE_ASYNC_LDS
  __builtin_amdgcn_global_load_async_to_lds_b128(
      (__attribute__((address_space(1))) v4i_t*)g,
      (__attribute__((address_space(3))) v4i_t*)l, 0, 0);
#else
  *(B128*)l = *(const B128*)g;
#endif
}

__device__ inline void cp_drain() {
#if HAVE_ASYNC_LDS
  __builtin_amdgcn_s_wait_asynccnt(0);
#endif
}

// ---------------------------------------------------------------------------
// 1) modulation: out[b, 0:6H] = silu(vec[b,:]) @ W + bias
// ---------------------------------------------------------------------------
__global__ __launch_bounds__(256)
void mod_kernel(const float* __restrict__ vec, const float* __restrict__ W,
                const float* __restrict__ bias, float* __restrict__ out)
{
  const int N6 = 6 * H_;
  int b = blockIdx.y;
  int n = blockIdx.x * 256 + threadIdx.x;
  __shared__ float sv[H_];
  for (int i = threadIdx.x; i < H_; i += 256) {
    float v = vec[(size_t)b * H_ + i];
    sv[i] = v / (1.f + expf(-v));
  }
  __syncthreads();
  float acc = bias[n];
  for (int k = 0; k < H_; ++k)
    acc = fmaf(sv[k], W[(size_t)k * N6 + n], acc);
  out[(size_t)b * N6 + n] = acc;
}

// ---------------------------------------------------------------------------
// 2) LayerNorm + modulate -> bf16:  out = LN(x)*(1+scale)+shift
// ---------------------------------------------------------------------------
__global__ __launch_bounds__(256)
void ln_mod_kernel(const float* __restrict__ X, const float* __restrict__ mod,
                   int shiftOff, int scaleOff, int rowsPerBatch,
                   unsigned short* __restrict__ Out)
{
  int row = blockIdx.x;
  int b   = row / rowsPerBatch;
  const float* x    = X + (size_t)row * H_;
  const float* mrow = mod + (size_t)b * (6 * H_);
  float s = 0.f, s2 = 0.f;
  for (int i = threadIdx.x; i < H_; i += 256) {
    float v = x[i];
    s += v; s2 += v * v;
  }
  __shared__ float sh[8];
  int lane = threadIdx.x & 31, w = threadIdx.x >> 5;
  #pragma unroll
  for (int o = 16; o > 0; o >>= 1) s += __shfl_down(s, o);
  if (lane == 0) sh[w] = s;
  __syncthreads();
  if (threadIdx.x == 0) { float t = 0.f; for (int i = 0; i < 8; ++i) t += sh[i]; sh[0] = t; }
  __syncthreads();
  float mean = sh[0] / (float)H_;
  __syncthreads();
  #pragma unroll
  for (int o = 16; o > 0; o >>= 1) s2 += __shfl_down(s2, o);
  if (lane == 0) sh[w] = s2;
  __syncthreads();
  if (threadIdx.x == 0) { float t = 0.f; for (int i = 0; i < 8; ++i) t += sh[i]; sh[0] = t; }
  __syncthreads();
  float var  = sh[0] / (float)H_ - mean * mean;
  float rstd = rsqrtf(var + 1e-6f);
  unsigned short* orow = Out + (size_t)row * H_;
  for (int i = threadIdx.x; i < H_; i += 256) {
    float xm = (x[i] - mean) * rstd;
    float y  = xm * (1.f + mrow[scaleOff + i]) + mrow[shiftOff + i];
    orow[i]  = f32_to_bf16(y);
  }
}

// ---------------------------------------------------------------------------
// 3) tiled bf16 WMMA GEMM: C[M,N] = A(bf16,MxK) @ W(f32,KxN) + bias, epilogue
//    block = 256 threads = 8 waves; tile 128x128x32, double-buffered LDS;
//    each wave computes a 32x64 sub-tile = 8 wmma per K-step.
// ---------------------------------------------------------------------------
enum { EPI_F32 = 0, EPI_GELU = 1, EPI_RESGATE = 2, EPI_RESGATE_CLIP = 3 };

template <int EPI>
__global__ __launch_bounds__(256)
void gemm_kernel(const unsigned short* __restrict__ A,
                 const float* __restrict__ W,
                 const float* __restrict__ bias,
                 float* __restrict__ outF,
                 unsigned short* __restrict__ outB,
                 int M, int N, int K,
                 const float* __restrict__ res,
                 const float* __restrict__ gate,
                 int rowsPerBatch, int gateStride)
{
  constexpr int BK = 32;
  __shared__ unsigned short As[2][128 * BK];   // A tile, row-major bf16
  __shared__ unsigned short Bs[2][128 * BK];   // W tile transposed Bs[n][k]

  int tid  = threadIdx.x;
  int lane = tid & 31;
  int wave = tid >> 5;                 // 0..7
  int bm   = blockIdx.y * 128;
  int bn   = blockIdx.x * 128;
  int wm   = (wave & 3) * 32;          // 4 wave-rows
  int wn   = (wave >> 2) * 64;         // 2 wave-cols

  floatx8 acc[2][4];
  #pragma unroll
  for (int mi = 0; mi < 2; ++mi)
    #pragma unroll
    for (int ni = 0; ni < 4; ++ni) acc[mi][ni] = {};

  auto load_tile = [&](int k0, int buf) {
    // A tile: 128x32 bf16, 32 contiguous bytes per thread (async when avail)
    {
      int r = tid >> 1;
      int c = (tid & 1) * 16;
      const unsigned short* src = A + (size_t)(bm + r) * K + k0 + c;
      cp16_g2l(src,     &As[buf][r * BK + c]);
      cp16_g2l(src + 8, &As[buf][r * BK + c + 8]);
    }
    // W tile: 32x128 f32 -> cvt bf16 -> transposed Bs[n][k]
    {
      int n   = tid & 127;
      int kk0 = (tid >> 7) * 16;
      const float* src = W + (size_t)(k0 + kk0) * N + bn + n;
      unsigned short* dst = &Bs[buf][n * BK + kk0];
      if (k0 + BK < K)
        __builtin_prefetch(W + (size_t)(k0 + BK + kk0) * N + bn + n, 0, 1);
      #pragma unroll
      for (int i = 0; i < 16; ++i)
        dst[i] = f32_to_bf16(src[(size_t)i * N]);
    }
    cp_drain();
  };

  load_tile(0, 0);
  int buf = 0;
  int kb = (lane >> 4) * 8;

  for (int k0 = 0; k0 < K; k0 += BK) {
    __syncthreads();                       // tile[buf] visible to all waves
    if (k0 + BK < K) load_tile(k0 + BK, buf ^ 1);

    Frag a[2], b[4];
    #pragma unroll
    for (int mi = 0; mi < 2; ++mi) {
      int r = wm + mi * 16 + (lane & 15);
      a[mi].q[0] = *(const B128*)&As[buf][r * BK + kb];
      a[mi].q[1] = *(const B128*)&As[buf][r * BK + kb + 16];
    }
    #pragma unroll
    for (int ni = 0; ni < 4; ++ni) {
      int c = wn + ni * 16 + (lane & 15);
      b[ni].q[0] = *(const B128*)&Bs[buf][c * BK + kb];
      b[ni].q[1] = *(const B128*)&Bs[buf][c * BK + kb + 16];
    }
    #pragma unroll
    for (int mi = 0; mi < 2; ++mi)
      #pragma unroll
      for (int ni = 0; ni < 4; ++ni)
        acc[mi][ni] = wmma_bf16(a[mi], b[ni], acc[mi][ni]);

    buf ^= 1;
  }

  // epilogue; C layout: row = tile + r + 8*(lane>>4), col = tile + (lane&15)
  #pragma unroll
  for (int mi = 0; mi < 2; ++mi) {
    #pragma unroll
    for (int ni = 0; ni < 4; ++ni) {
      #pragma unroll
      for (int r = 0; r < 8; ++r) {
        int row = bm + wm + mi * 16 + (lane >> 4) * 8 + r;
        int col = bn + wn + ni * 16 + (lane & 15);
        float x = acc[mi][ni][r] + bias[col];
        size_t o = (size_t)row * N + col;
        if (EPI == EPI_F32) {
          outF[o] = x;
        } else if (EPI == EPI_GELU) {
          float g = 0.5f * x * (1.f + tanhf(0.7978845608028654f * (x + 0.044715f * x * x * x)));
          outB[o] = f32_to_bf16(g);
        } else {
          int bb  = row / rowsPerBatch;
          float y = res[o] + x * gate[(size_t)bb * gateStride + col];
          if (EPI == EPI_RESGATE_CLIP) y = fminf(fmaxf(y, -65504.f), 65504.f);
          outF[o] = y;
        }
      }
    }
  }
}

// ---------------------------------------------------------------------------
// 4) qkv prep: RMSNorm(q,k) + RoPE, scatter to bf16 (b,h,l,d) buffers
// ---------------------------------------------------------------------------
__global__ __launch_bounds__(128)
void qkv_prep_kernel(const float* __restrict__ qkv, int Lx, int lbase,
                     const float* __restrict__ qscale, const float* __restrict__ kscale,
                     const float* __restrict__ pe,
                     unsigned short* __restrict__ Qb, unsigned short* __restrict__ Kb,
                     unsigned short* __restrict__ Vb)
{
  int l = blockIdx.x, h = blockIdx.y, b = blockIdx.z;
  int t = threadIdx.x;                 // 0..127 == head dim index
  const float* rowp = qkv + ((size_t)b * Lx + l) * (3 * H_);
  float q = rowp[h * DH_ + t];
  float k = rowp[H_ + h * DH_ + t];
  float v = rowp[2 * H_ + h * DH_ + t];

  __shared__ float sq[4], sk[4];
  float aq = q * q, ak = k * k;
  #pragma unroll
  for (int o = 16; o > 0; o >>= 1) { aq += __shfl_down(aq, o); ak += __shfl_down(ak, o); }
  int lane = t & 31, w = t >> 5;
  if (lane == 0) { sq[w] = aq; sk[w] = ak; }
  __syncthreads();
  float tq = sq[0] + sq[1] + sq[2] + sq[3];
  float tk = sk[0] + sk[1] + sk[2] + sk[3];
  float qn = q * rsqrtf(tq / (float)DH_ + 1e-6f) * qscale[t];
  float kn = k * rsqrtf(tk / (float)DH_ + 1e-6f) * kscale[t];

  // RoPE: pair (2i, 2i+1); partner lives in adjacent lane (wave32 shfl_xor)
  int lg = lbase + l;
  int i  = t >> 1, j = t & 1;
  const float* pp = pe + (((size_t)lg * (DH_ / 2) + i) * 2 + j) * 2;
  float qo = __shfl_xor(qn, 1), ko = __shfl_xor(kn, 1);
  float q0 = j ? qo : qn, q1 = j ? qn : qo;
  float k0 = j ? ko : kn, k1 = j ? kn : ko;
  float qr = pp[0] * q0 + pp[1] * q1;
  float kr = pp[0] * k0 + pp[1] * k1;

  size_t off = (((size_t)b * NH_ + h) * Ltot_ + lg) * DH_ + t;
  Qb[off] = f32_to_bf16(qr);
  Kb[off] = f32_to_bf16(kr);
  Vb[off] = f32_to_bf16(v);
}

// ---------------------------------------------------------------------------
// 5) flash attention: 64-row Q tile per block, 64-key KV tiles, WMMA S & PV
// ---------------------------------------------------------------------------
__global__ __launch_bounds__(128)
void attn_kernel(const unsigned short* __restrict__ Qb,
                 const unsigned short* __restrict__ Kb,
                 const unsigned short* __restrict__ Vb,
                 unsigned short* __restrict__ attn_txt,
                 unsigned short* __restrict__ attn_img)
{
  __shared__ unsigned short Qs[64 * 128];   // 16 KB
  __shared__ unsigned short Ks[64 * 128];   // 16 KB
  __shared__ unsigned short Vt[128 * 64];   // 16 KB, transposed: Vt[d][kv]
  __shared__ unsigned short Ps[64 * 64];    //  8 KB, probs bf16

  int qt = blockIdx.x, h = blockIdx.y, b = blockIdx.z;
  int tid = threadIdx.x, lane = tid & 31, wave = tid >> 5;
  const size_t headOff = ((size_t)b * NH_ + h) * (size_t)Ltot_ * DH_;

  // Q tile is contiguous in (b,h,l,d) layout -> async copy
  for (int i = tid; i < 1024; i += 128)
    cp16_g2l(&Qb[headOff + (size_t)qt * 64 * 128 + i * 8], &Qs[i * 8]);
  cp_drain();

  float m_i[8], l_i[8];
  floatx8 o_acc[8];
  #pragma unroll
  for (int r = 0; r < 8; ++r) { m_i[r] = -1e30f; l_i[r] = 0.f; }
  #pragma unroll
  for (int ni = 0; ni < 8; ++ni) o_acc[ni] = {};

  const float scl = 0.08838834764831845f;   // 1/sqrt(128)
  int rA = wave * 16 + (lane & 15);
  int kb = (lane >> 4) * 8;

  for (int kt = 0; kt < Ltot_ / 64; ++kt) {
    __syncthreads();
    // K tile contiguous (async); V tile transposed into Vt
    for (int i = tid; i < 1024; i += 128)
      cp16_g2l(&Kb[headOff + (size_t)kt * 64 * 128 + i * 8], &Ks[i * 8]);
    for (int i = tid; i < 1024; i += 128) {
      int r = i >> 4;            // kv row
      int c = (i & 15) * 8;      // d col
      B128 d = *(const B128*)&Vb[headOff + (size_t)(kt * 64 + r) * 128 + c];
      const unsigned short* u = (const unsigned short*)&d;
      #pragma unroll
      for (int j = 0; j < 8; ++j) Vt[(c + j) * 64 + r] = u[j];
    }
    cp_drain();
    __syncthreads();

    // --- S = Q K^T over this kv tile (wave: 16 rows x 64 keys)
    Frag aq[4];
    #pragma unroll
    for (int kk = 0; kk < 4; ++kk) {
      aq[kk].q[0] = *(const B128*)&Qs[rA * 128 + kk * 32 + kb];
      aq[kk].q[1] = *(const B128*)&Qs[rA * 128 + kk * 32 + kb + 16];
    }
    floatx8 s[4];
    #pragma unroll
    for (int ni = 0; ni < 4; ++ni) s[ni] = {};
    #pragma unroll
    for (int kk = 0; kk < 4; ++kk) {
      #pragma unroll
      for (int ni = 0; ni < 4; ++ni) {
        Frag bk;
        int key = ni * 16 + (lane & 15);
        bk.q[0] = *(const B128*)&Ks[key * 128 + kk * 32 + kb];
        bk.q[1] = *(const B128*)&Ks[key * 128 + kk * 32 + kb + 16];
        s[ni] = wmma_bf16(aq[kk], bk, s[ni]);
      }
    }

    // --- online softmax (row reductions across the 16-lane column group)
    #pragma unroll
    for (int r = 0; r < 8; ++r) {
      float mx = -1e30f;
      #pragma unroll
      for (int ni = 0; ni < 4; ++ni) { s[ni][r] *= scl; mx = fmaxf(mx, s[ni][r]); }
      #pragma unroll
      for (int o = 1; o < 16; o <<= 1) mx = fmaxf(mx, __shfl_xor(mx, o));
      float nm = fmaxf(m_i[r], mx);
      float f  = expf(m_i[r] - nm);
      float rs = 0.f;
      #pragma unroll
      for (int ni = 0; ni < 4; ++ni) {
        float p = expf(s[ni][r] - nm);
        s[ni][r] = p;
        rs += p;
      }
      #pragma unroll
      for (int o = 1; o < 16; o <<= 1) rs += __shfl_xor(rs, o);
      l_i[r] = l_i[r] * f + rs;
      m_i[r] = nm;
      #pragma unroll
      for (int ni = 0; ni < 8; ++ni) o_acc[ni][r] *= f;
    }

    // --- stash P as bf16 in A-matrix-loadable layout
    int prow = wave * 16 + (lane >> 4) * 8;
    #pragma unroll
    for (int ni = 0; ni < 4; ++ni) {
      int col = ni * 16 + (lane & 15);
      #pragma unroll
      for (int r = 0; r < 8; ++r)
        Ps[(prow + r) * 64 + col] = f32_to_bf16(s[ni][r]);
    }
    __syncthreads();

    // --- O += P @ V  (wave: 16 rows x 128 dh, K = 64)
    Frag ap[2];
    #pragma unroll
    for (int kk = 0; kk < 2; ++kk) {
      ap[kk].q[0] = *(const B128*)&Ps[rA * 64 + kk * 32 + kb];
      ap[kk].q[1] = *(const B128*)&Ps[rA * 64 + kk * 32 + kb + 16];
    }
    #pragma unroll
    for (int ni = 0; ni < 8; ++ni) {
      int d = ni * 16 + (lane & 15);
      #pragma unroll
      for (int kk = 0; kk < 2; ++kk) {
        Frag bv;
        bv.q[0] = *(const B128*)&Vt[d * 64 + kk * 32 + kb];
        bv.q[1] = *(const B128*)&Vt[d * 64 + kk * 32 + kb + 16];
        o_acc[ni] = wmma_bf16(ap[kk], bv, o_acc[ni]);
      }
    }
  }

  // --- finalize: O /= l, scatter to txt/img attn buffers (bf16, row-major H)
  #pragma unroll
  for (int r = 0; r < 8; ++r) {
    int lg = qt * 64 + wave * 16 + (lane >> 4) * 8 + r;
    float inv = 1.f / l_i[r];
    bool isTxt = lg < LT_;
    size_t rowOff = isTxt ? ((size_t)b * LT_ + lg) * H_
                          : ((size_t)b * LI_ + (lg - LT_)) * H_;
    unsigned short* base = isTxt ? attn_txt : attn_img;
    #pragma unroll
    for (int ni = 0; ni < 8; ++ni) {
      int col = h * DH_ + ni * 16 + (lane & 15);
      base[rowOff + col] = f32_to_bf16(o_acc[ni][r] * inv);
    }
  }
}

// ---------------------------------------------------------------------------
// launcher
// ---------------------------------------------------------------------------
extern "C" void kernel_launch(void* const* d_in, const int* in_sizes, int n_in,
                              void* d_out, int out_size, void* d_ws, size_t ws_size,
                              hipStream_t stream)
{
  const float* img        = (const float*)d_in[0];
  const float* txt        = (const float*)d_in[1];
  const float* vec        = (const float*)d_in[2];
  const float* pe         = (const float*)d_in[3];
  const float* img_mod_w  = (const float*)d_in[4];
  const float* img_mod_b  = (const float*)d_in[5];
  const float* txt_mod_w  = (const float*)d_in[6];
  const float* txt_mod_b  = (const float*)d_in[7];
  const float* img_qkv_w  = (const float*)d_in[8];
  const float* img_qkv_b  = (const float*)d_in[9];
  const float* img_qs     = (const float*)d_in[10];
  const float* img_ks     = (const float*)d_in[11];
  const float* img_proj_w = (const float*)d_in[12];
  const float* img_proj_b = (const float*)d_in[13];
  const float* img_mlp_w1 = (const float*)d_in[14];
  const float* img_mlp_b1 = (const float*)d_in[15];
  const float* img_mlp_w2 = (const float*)d_in[16];
  const float* img_mlp_b2 = (const float*)d_in[17];
  const float* txt_qkv_w  = (const float*)d_in[18];
  const float* txt_qkv_b  = (const float*)d_in[19];
  const float* txt_qs     = (const float*)d_in[20];
  const float* txt_ks     = (const float*)d_in[21];
  const float* txt_proj_w = (const float*)d_in[22];
  const float* txt_proj_b = (const float*)d_in[23];
  const float* txt_mlp_w1 = (const float*)d_in[24];
  const float* txt_mlp_b1 = (const float*)d_in[25];
  const float* txt_mlp_w2 = (const float*)d_in[26];
  const float* txt_mlp_b2 = (const float*)d_in[27];

  float* out_img = (float*)d_out;                              // (B,LI,H)
  float* out_txt = (float*)d_out + (size_t)B_ * LI_ * H_;      // (B,LT,H)

  const int Mimg = B_ * LI_;     // 2048
  const int Mtxt = B_ * LT_;     // 512

  char* wsp = (char*)d_ws;
  size_t off = 0;
  auto alloc = [&](size_t bytes) -> void* {
    void* p = wsp + off;
    off += (bytes + 255) & ~(size_t)255;
    return p;
  };

  float* imod = (float*)alloc((size_t)B_ * 6 * H_ * 4);
  float* tmod = (float*)alloc((size_t)B_ * 6 * H_ * 4);
  unsigned short* xm_img = (unsigned short*)alloc((size_t)Mimg * H_ * 2);
  unsigned short* xm_txt = (unsigned short*)alloc((size_t)Mtxt * H_ * 2);
  float* qkv_img = (float*)alloc((size_t)Mimg * 3 * H_ * 4);
  float* qkv_txt = (float*)alloc((size_t)Mtxt * 3 * H_ * 4);
  size_t qkvElems = (size_t)B_ * NH_ * Ltot_ * DH_;
  unsigned short* Qb = (unsigned short*)alloc(qkvElems * 2);
  unsigned short* Kb = (unsigned short*)alloc(qkvElems * 2);
  unsigned short* Vb = (unsigned short*)alloc(qkvElems * 2);
  unsigned short* attn_img = (unsigned short*)alloc((size_t)Mimg * H_ * 2);
  unsigned short* attn_txt = (unsigned short*)alloc((size_t)Mtxt * H_ * 2);
  float* img1 = (float*)alloc((size_t)Mimg * H_ * 4);
  float* txt1 = (float*)alloc((size_t)Mtxt * H_ * 4);
  unsigned short* hm_img = (unsigned short*)alloc((size_t)Mimg * H_ * 2);
  unsigned short* hm_txt = (unsigned short*)alloc((size_t)Mtxt * H_ * 2);
  unsigned short* g_img  = (unsigned short*)alloc((size_t)Mimg * MLP_ * 2);
  unsigned short* g_txt  = (unsigned short*)alloc((size_t)Mtxt * MLP_ * 2);

  // 1) modulation vectors
  mod_kernel<<<dim3(6 * H_ / 256, B_), 256, 0, stream>>>(vec, img_mod_w, img_mod_b, imod);
  mod_kernel<<<dim3(6 * H_ / 256, B_), 256, 0, stream>>>(vec, txt_mod_w, txt_mod_b, tmod);

  // 2) LN + mod1 -> bf16
  ln_mod_kernel<<<Mimg, 256, 0, stream>>>(img, imod, 0, H_, LI_, xm_img);
  ln_mod_kernel<<<Mtxt, 256, 0, stream>>>(txt, tmod, 0, H_, LT_, xm_txt);

  // 3) QKV GEMMs (f32 out)
  gemm_kernel<EPI_F32><<<dim3(3 * H_ / 128, Mimg / 128), 256, 0, stream>>>(
      xm_img, img_qkv_w, img_qkv_b, qkv_img, nullptr, Mimg, 3 * H_, H_,
      nullptr, nullptr, 1, 0);
  gemm_kernel<EPI_F32><<<dim3(3 * H_ / 128, Mtxt / 128), 256, 0, stream>>>(
      xm_txt, txt_qkv_w, txt_qkv_b, qkv_txt, nullptr, Mtxt, 3 * H_, H_,
      nullptr, nullptr, 1, 0);

  // 4) RMSNorm + RoPE + layout change (txt occupies seq [0,256), img [256,1280))
  qkv_prep_kernel<<<dim3(LT_, NH_, B_), 128, 0, stream>>>(
      qkv_txt, LT_, 0, txt_qs, txt_ks, pe, Qb, Kb, Vb);
  qkv_prep_kernel<<<dim3(LI_, NH_, B_), 128, 0, stream>>>(
      qkv_img, LI_, LT_, img_qs, img_ks, pe, Qb, Kb, Vb);

  // 5) attention
  attn_kernel<<<dim3(Ltot_ / 64, NH_, B_), 128, 0, stream>>>(
      Qb, Kb, Vb, attn_txt, attn_img);

  // 6) proj + residual*gate1
  gemm_kernel<EPI_RESGATE><<<dim3(H_ / 128, Mimg / 128), 256, 0, stream>>>(
      attn_img, img_proj_w, img_proj_b, img1, nullptr, Mimg, H_, H_,
      img, imod + 2 * H_, LI_, 6 * H_);
  gemm_kernel<EPI_RESGATE><<<dim3(H_ / 128, Mtxt / 128), 256, 0, stream>>>(
      attn_txt, txt_proj_w, txt_proj_b, txt1, nullptr, Mtxt, H_, H_,
      txt, tmod + 2 * H_, LT_, 6 * H_);

  // 7) LN + mod2 -> bf16
  ln_mod_kernel<<<Mimg, 256, 0, stream>>>(img1, imod, 3 * H_, 4 * H_, LI_, hm_img);
  ln_mod_kernel<<<Mtxt, 256, 0, stream>>>(txt1, tmod, 3 * H_, 4 * H_, LT_, hm_txt);

  // 8) MLP up + GELU -> bf16
  gemm_kernel<EPI_GELU><<<dim3(MLP_ / 128, Mimg / 128), 256, 0, stream>>>(
      hm_img, img_mlp_w1, img_mlp_b1, nullptr, g_img, Mimg, MLP_, H_,
      nullptr, nullptr, 1, 0);
  gemm_kernel<EPI_GELU><<<dim3(MLP_ / 128, Mtxt / 128), 256, 0, stream>>>(
      hm_txt, txt_mlp_w1, txt_mlp_b1, nullptr, g_txt, Mtxt, MLP_, H_,
      nullptr, nullptr, 1, 0);

  // 9) MLP down + residual*gate2 -> final outputs (txt clipped)
  gemm_kernel<EPI_RESGATE><<<dim3(H_ / 128, Mimg / 128), 256, 0, stream>>>(
      g_img, img_mlp_w2, img_mlp_b2, out_img, nullptr, Mimg, H_, MLP_,
      img1, imod + 5 * H_, LI_, 6 * H_);
  gemm_kernel<EPI_RESGATE_CLIP><<<dim3(H_ / 128, Mtxt / 128), 256, 0, stream>>>(
      g_txt, txt_mlp_w2, txt_mlp_b2, out_txt, nullptr, Mtxt, H_, MLP_,
      txt1, tmod + 5 * H_, LT_, 6 * H_);

  (void)in_sizes; (void)n_in; (void)out_size; (void)ws_size;
}
